// Decoder_bipartite_81071802679526
// MI455X (gfx1250) — matrix operand order repaired and checked by reading, not compiled, not run on hardware
//
#include <hip/hip_runtime.h>
#include <hip/hip_bf16.h>

typedef __attribute__((ext_vector_type(16))) __bf16 v16bf;
typedef __attribute__((ext_vector_type(8)))  float  v8f;

#define HDIM 256
#define TWO_H 512
#define KSTEPS 16          // 512 / 32
#define NTILES 16          // 256 / 16
#define NT_PER_WAVE 8      // N-tiles per wave (N split across wave pairs)
#define EDGES_PER_BLOCK 128
#define THREADS 256
#define W1B_ELEMS (KSTEPS * NTILES * 32 * 16)   // 131072 bf16 = 256 KB

// ---------------------------------------------------------------------------
// Prep: convert w1 (f32 [256][512], row-major, out x in) into bf16 fragments.
// ws layout: [kt][nt][lane][j]  (j = 0..15 bf16 per lane)
//   B-matrix (K x N) wave32 layout: lane L holds column n = L%16,
//   K values k = kt*32 + (j/8)*16 + (L/16)*8 + j%8, and B[k][n] = w1[n][k].
// ---------------------------------------------------------------------------
__global__ void prep_w1_bf16(const float* __restrict__ w1, __bf16* __restrict__ wsb) {
    int t = blockIdx.x * blockDim.x + threadIdx.x;
    if (t >= W1B_ELEMS) return;
    int j  =  t        & 15;
    int L  = (t >> 4)  & 31;
    int nt = (t >> 9)  & 15;
    int kt =  t >> 13;
    int n  = nt * 16 + (L & 15);
    int k  = kt * 32 + (j >> 3) * 16 + (L >> 4) * 8 + (j & 7);
    wsb[t] = (__bf16)w1[n * TWO_H + k];
}

__device__ __forceinline__ v16bf cvt_frag(float4 f0, float4 f1, float4 f2, float4 f3) {
    v16bf a;
    a[0]  = (__bf16)f0.x;  a[1]  = (__bf16)f0.y;
    a[2]  = (__bf16)f0.z;  a[3]  = (__bf16)f0.w;
    a[4]  = (__bf16)f1.x;  a[5]  = (__bf16)f1.y;
    a[6]  = (__bf16)f1.z;  a[7]  = (__bf16)f1.w;
    a[8]  = (__bf16)f2.x;  a[9]  = (__bf16)f2.y;
    a[10] = (__bf16)f2.z;  a[11] = (__bf16)f2.w;
    a[12] = (__bf16)f3.x;  a[13] = (__bf16)f3.y;
    a[14] = (__bf16)f3.z;  a[15] = (__bf16)f3.w;
    return a;
}

// ---------------------------------------------------------------------------
// Fused decoder. Each wave: 2 M-tiles (32 edges) x 8 N-tiles, so every B
// fragment loaded from LDS feeds two WMMAs (1:1 wmma:ds_load_b128 ratio).
// N dimension split across wave pairs; partials combined through LDS.
// ---------------------------------------------------------------------------
__global__ void __launch_bounds__(THREADS, 1)
decoder_kernel(const float* __restrict__ z_src,
               const float* __restrict__ z_dst,
               const long long* __restrict__ eli,   // [2][E] int64
               const float* __restrict__ b1,
               const float* __restrict__ w2,
               const float* __restrict__ b2,
               const __bf16* __restrict__ w1b,
               float* __restrict__ out,
               int E) {
    extern __shared__ __align__(32) __bf16 ldsB[];       // 256 KB weights
    float* ldsPartial = (float*)(ldsB + W1B_ELEMS);      // 128 floats

    const int tid  = threadIdx.x;
    const int wave = tid >> 5;
    const int lane = tid & 31;
    const int l16  = lane & 15;
    const int hlf  = lane >> 4;
    const int mg   = wave & 3;    // edge group (4 x 32 edges)
    const int nh   = wave >> 2;   // n-half (0: n<128, 1: n>=128)
    const int groupBase = blockIdx.x * EDGES_PER_BLOCK + mg * 32;

    // ---- stage ALL B fragments into LDS (one-time) ------------------------
    {
        const uint4* __restrict__ src = (const uint4*)w1b;
        uint4* dst = (uint4*)ldsB;
        #pragma unroll 4
        for (int i = tid; i < W1B_ELEMS / 8; i += THREADS)   // 16384 uint4
            dst[i] = src[i];
    }

    // ---- per-lane gather pointers for the wave's two 16-edge A tiles ------
    int e0 = groupBase + l16;       if (e0 > E - 1) e0 = E - 1;
    int e1 = groupBase + 16 + l16;  if (e1 > E - 1) e1 = E - 1;
    const long long rs0 = eli[e0], cd0 = eli[(long long)E + e0];
    const long long rs1 = eli[e1], cd1 = eli[(long long)E + e1];
    const float* __restrict__ psrc0 = z_src + rs0 * (long long)HDIM;
    const float* __restrict__ pdst0 = z_dst + cd0 * (long long)HDIM;
    const float* __restrict__ psrc1 = z_src + rs1 * (long long)HDIM;
    const float* __restrict__ pdst1 = z_dst + cd1 * (long long)HDIM;

    v8f C[2][NT_PER_WAVE] = {};   // 128 VGPRs of accumulators

    // Prefetch raw A floats for kt = 0 (uniform: kt<8 -> src half).
    float4 fa0, fa1, fa2, fa3, fb0, fb1, fb2, fb3;
    {
        const float* p0 = psrc0 + hlf * 8;
        const float* p1 = psrc0 + 16 + hlf * 8;
        const float* q0 = psrc1 + hlf * 8;
        const float* q1 = psrc1 + 16 + hlf * 8;
        fa0 = ((const float4*)p0)[0];  fa1 = ((const float4*)p0)[1];
        fa2 = ((const float4*)p1)[0];  fa3 = ((const float4*)p1)[1];
        fb0 = ((const float4*)q0)[0];  fb1 = ((const float4*)q0)[1];
        fb2 = ((const float4*)q1)[0];  fb3 = ((const float4*)q1)[1];
    }

    __syncthreads();   // weights staged

    #pragma unroll
    for (int kt = 0; kt < KSTEPS; ++kt) {
        // A fragments (16-bit A 16x32: element j -> k = (j/8)*16 + (L/16)*8 + j%8)
        v16bf a0 = cvt_frag(fa0, fa1, fa2, fa3);
        v16bf a1 = cvt_frag(fb0, fb1, fb2, fb3);

        // Prefetch next K-step's raw floats (overlaps with WMMA burst).
        if (kt < KSTEPS - 1) {
            const int ktn = kt + 1;
            const float* b0 = ((ktn < 8) ? psrc0 : pdst0) + (ktn & 7) * 32;
            const float* b1p = ((ktn < 8) ? psrc1 : pdst1) + (ktn & 7) * 32;
            const float* p0 = b0 + hlf * 8;
            const float* p1 = b0 + 16 + hlf * 8;
            const float* q0 = b1p + hlf * 8;
            const float* q1 = b1p + 16 + hlf * 8;
            fa0 = ((const float4*)p0)[0];  fa1 = ((const float4*)p0)[1];
            fa2 = ((const float4*)p1)[0];  fa3 = ((const float4*)p1)[1];
            fb0 = ((const float4*)q0)[0];  fb1 = ((const float4*)q0)[1];
            fb2 = ((const float4*)q1)[0];  fb3 = ((const float4*)q1)[1];
        }

        // B fragments for this wave's 8 N-tiles; each feeds two WMMAs.
        const __bf16* ktBase = ldsB + ((kt * NTILES + nh * NT_PER_WAVE) * 32 + lane) * 16;
        v16bf bcur = *(const v16bf*)(ktBase);
        #pragma unroll
        for (int j = 0; j < NT_PER_WAVE; ++j) {
            v16bf bnext;
            if (j < NT_PER_WAVE - 1)
                bnext = *(const v16bf*)(ktBase + (j + 1) * 32 * 16);
            C[0][j] = __builtin_amdgcn_wmma_f32_16x16x32_bf16(
                          false, a0, false, bcur, (short)0, C[0][j], false, false);
            C[1][j] = __builtin_amdgcn_wmma_f32_16x16x32_bf16(
                          false, a1, false, bcur, (short)0, C[1][j], false, false);
            bcur = bnext;
        }
    }

    // ---- Fused layer 2: relu(C + b1) . w2 over this wave's n-half ---------
    // C/D layout: VGPR r, lane L -> row M = (L/16)*8 + r, col N = L%16.
    float acc[2][8];
    #pragma unroll
    for (int t = 0; t < 2; ++t)
        #pragma unroll
        for (int r = 0; r < 8; ++r) acc[t][r] = 0.f;

    #pragma unroll
    for (int j = 0; j < NT_PER_WAVE; ++j) {
        const int n = nh * 128 + j * 16 + l16;
        const float b1v = b1[n];
        const float w2v = w2[n];
        #pragma unroll
        for (int t = 0; t < 2; ++t)
            #pragma unroll
            for (int r = 0; r < 8; ++r) {
                float c = C[t][j][r] + b1v;
                c = c > 0.f ? c : 0.f;
                acc[t][r] += c * w2v;
            }
    }

    // Reduce across the 16 lanes of each column group.
    #pragma unroll
    for (int m = 1; m < 16; m <<= 1)
        #pragma unroll
        for (int t = 0; t < 2; ++t)
            #pragma unroll
            for (int r = 0; r < 8; ++r)
                acc[t][r] += __shfl_xor(acc[t][r], m, 32);

    // Per tile t, lanes l16<8 hold the sum for local row hlf*8 + l16.
    float v0 = acc[0][0], v1 = acc[1][0];
    if (l16 == 1) { v0 = acc[0][1]; v1 = acc[1][1]; }
    if (l16 == 2) { v0 = acc[0][2]; v1 = acc[1][2]; }
    if (l16 == 3) { v0 = acc[0][3]; v1 = acc[1][3]; }
    if (l16 == 4) { v0 = acc[0][4]; v1 = acc[1][4]; }
    if (l16 == 5) { v0 = acc[0][5]; v1 = acc[1][5]; }
    if (l16 == 6) { v0 = acc[0][6]; v1 = acc[1][6]; }
    if (l16 == 7) { v0 = acc[0][7]; v1 = acc[1][7]; }

    const int loc0 = mg * 32 + hlf * 8 + l16;        // local edge, tile 0
    const int loc1 = loc0 + 16;                      // local edge, tile 1

    if (nh == 0 && l16 < 8) {
        ldsPartial[loc0] = v0;
        ldsPartial[loc1] = v1;
    }
    __syncthreads();
    if (nh == 1 && l16 < 8) {
        const int base = blockIdx.x * EDGES_PER_BLOCK;
        const float x0 = v0 + ldsPartial[loc0] + b2[0];
        const float x1 = v1 + ldsPartial[loc1] + b2[0];
        if (base + loc0 < E) out[base + loc0] = 1.f / (1.f + __expf(-x0));
        if (base + loc1 < E) out[base + loc1] = 1.f / (1.f + __expf(-x1));
    }
}

// ---------------------------------------------------------------------------
extern "C" void kernel_launch(void* const* d_in, const int* in_sizes, int n_in,
                              void* d_out, int out_size, void* d_ws, size_t ws_size,
                              hipStream_t stream) {
    const float*     z_src = (const float*)d_in[0];
    const float*     z_dst = (const float*)d_in[1];
    const long long* eli   = (const long long*)d_in[2];
    const float*     w1    = (const float*)d_in[3];
    const float*     b1    = (const float*)d_in[4];
    const float*     w2    = (const float*)d_in[5];
    const float*     b2    = (const float*)d_in[6];
    float*           out   = (float*)d_out;
    const int E = in_sizes[2] / 2;

    __bf16* wsb = (__bf16*)d_ws;   // 256 KB needed

    prep_w1_bf16<<<(W1B_ELEMS + 255) / 256, 256, 0, stream>>>(w1, wsb);

    const int nblocks = (E + EDGES_PER_BLOCK - 1) / EDGES_PER_BLOCK;
    const size_t ldsBytes = (size_t)W1B_ELEMS * sizeof(__bf16)
                          + EDGES_PER_BLOCK * sizeof(float);   // 256 KB + 512 B
    decoder_kernel<<<nblocks, THREADS, ldsBytes, stream>>>(z_src, z_dst, eli,
                                                           b1, w2, b2, wsb, out, E);
}
